// HypConv2DPoincare_1803886264418
// MI455X (gfx1250) — matrix operand order repaired and checked by reading, not compiled, not run on hardware
//
#include <hip/hip_runtime.h>
#include <hip/hip_bf16.h>

typedef __attribute__((ext_vector_type(16))) _Float16 v16h;
typedef __attribute__((ext_vector_type(8)))  float    v8f;
typedef __attribute__((ext_vector_type(4)))  int      v4i;

#define NROWS (64*64*64)
#define KDIM  144
#define KPAD  160
#define NCOUT 32

#define HAS_ASYNC_LDS __has_builtin(__builtin_amdgcn_global_load_async_to_lds_b128)

// d_ws layout:
//   [0, 10240)           : z in f16, column-major, padded: zf[n*KPAD + k], k>=144 -> 0
//   [10240, +128)        : zn[32]   column norms (clamped to EPS)
//   [10368, +128)        : ch[32]   cosh(2*sqrt(c)*r)
//   [10496, +128)        : sh[32]   sinh(2*sqrt(c)*r)
//   [10624, +4)          : ratio    beta-concatenation scale
__global__ void hyp_prep(const float* __restrict__ z, const float* __restrict__ r, void* ws)
{
    _Float16* zf  = (_Float16*)ws;
    float* znp  = (float*)((char*)ws + NCOUT * KPAD * 2);
    float* chp  = znp + NCOUT;
    float* shp  = chp + NCOUT;
    float* ratp = shp + NCOUT;

    int t = threadIdx.x;
    if (t < NCOUT) {
        float ss = 0.0f;
        for (int k = 0; k < KPAD; ++k) {
            float v = (k < KDIM) ? z[k * NCOUT + t] : 0.0f;
            zf[t * KPAD + k] = (_Float16)v;
            ss += v * v;
        }
        znp[t] = fmaxf(sqrtf(ss), 1e-7f);
        float tc = 2.0f * r[t];          // sqrt(c) == 1
        chp[t] = coshf(tc);
        shp[t] = sinhf(tc);
    }
    if (t == 0) {
        // ratio = exp(logB(144/2,1/2) - logB(16/2,1/2))
        float lbn = lgammaf(72.0f) + lgammaf(0.5f) - lgammaf(72.5f);
        float lbi = lgammaf(8.0f)  + lgammaf(0.5f) - lgammaf(8.5f);
        *ratp = expf(lbn - lbi);
    }
}

__global__ __launch_bounds__(256) void hyp_conv(const float* __restrict__ x,
                                                float* __restrict__ out,
                                                const void* __restrict__ ws)
{
    // 4 clamped x rows (h0-1 .. h0+2) x 64 w x 16 ch, f32 = 16 KB
    __shared__ float tile[4 * 64 * 16];

    const _Float16* zf = (const _Float16*)ws;
    const float* znp = (const float*)((const char*)ws + NCOUT * KPAD * 2);
    const float* chp = znp + NCOUT;
    const float* shp = chp + NCOUT;
    const float  ratio = *(shp + NCOUT);

    const int tid      = threadIdx.x;
    const int blockRow = blockIdx.x << 7;          // 128 rows per block: h = h0, h0+1 (h0 even)
    const int b        = blockRow >> 12;
    const int h0       = (blockRow >> 6) & 63;
    const float* xb    = x + (size_t)b * (64u * 64u * 16u);

    // ---- stage halo tile into LDS (async-to-LDS when the toolchain exposes it) ----
    #pragma unroll
    for (int it = 0; it < 4; ++it) {
        int idx = tid + (it << 8);                 // float4 index, 0..1023
        int j   = idx >> 8;                        // LDS row 0..3
        int pos = idx & 255;                       // float4 within row (64*16/4)
        int g   = h0 - 1 + j; g = g < 0 ? 0 : (g > 63 ? 63 : g);
        const float4* src = (const float4*)(xb + (size_t)g * 1024) + pos;
#if HAS_ASYNC_LDS
        __builtin_amdgcn_global_load_async_to_lds_b128(
            (__attribute__((address_space(1))) v4i*)(const void*)src,
            (__attribute__((address_space(3))) v4i*)(((float4*)tile) + idx),
            0, 0);
#else
        ((float4*)tile)[idx] = *src;
#endif
    }
#if HAS_ASYNC_LDS
    asm volatile("s_wait_asynccnt 0" ::: "memory");
#endif
    __syncthreads();

    const int lane   = tid & 31;
    const int wave   = tid >> 5;
    const int rowBase = blockRow + (wave << 4);    // 16 rows per wave

    const int n0    = lane & 15;                   // col index / A-row-in-tile
    const int khalf = lane >> 4;                   // which K half this lane holds
    const int myRow = rowBase + n0;
    const int w     = myRow & 63;
    const int hsel  = wave >> 2;                   // 0 -> h0, 1 -> h0+1

    float ss = 0.0f;                               // partial patch sum-of-squares
    v8f acc0 = {};                                 // columns 0..15
    v8f acc1 = {};                                 // columns 16..31

    #pragma unroll
    for (int c = 0; c < 5; ++c) {
        // ---- A fragment (16x32 f16, contiguous k = 16*khalf + i) gathered from the LDS tile ----
        v16h a;
        #pragma unroll
        for (int i = 0; i < 16; ++i) {
            float val = 0.0f;
            if (!(c == 4 && khalf == 1)) {         // K>=144 only chunk4 lanes 16..31
                int k   = c * 32 + khalf * 16 + i;
                int dkh = k / 48;                  // patch order = (kh, kw, C)
                int rem = k - dkh * 48;
                int dkw = rem >> 4;
                int ci  = rem & 15;
                int jj  = hsel + dkh;              // clamp folded into staging
                int ww  = w - 1 + dkw; ww = ww < 0 ? 0 : (ww > 63 ? 63 : ww);
                val = tile[(((jj << 6) + ww) << 4) + ci];
                ss += val * val;
            }
            a[i] = (_Float16)val;
        }

        // ---- B fragments: 16 contiguous K halves per lane from column-major z_f16 ----
        const v16h* bp0 = (const v16h*)(zf + (size_t)(n0        * KPAD + c * 32 + khalf * 16));
        const v16h* bp1 = (const v16h*)(zf + (size_t)((16 + n0) * KPAD + c * 32 + khalf * 16));
        v16h b0 = *bp0;
        v16h b1 = *bp1;

        acc0 = __builtin_amdgcn_wmma_f32_16x16x32_f16(false, a, false, b0, (short)0, acc0, false, false);
        acc1 = __builtin_amdgcn_wmma_f32_16x16x32_f16(false, a, false, b1, (short)0, acc1, false, false);
    }

    // ---- per-row scalars (lane and lane^16 hold complementary halves of the row) ----
    float ssf    = ss + __shfl_xor(ss, 16);
    float yn2raw = ratio * ratio * ssf;
    float yn     = sqrtf(yn2raw);
    const float max_norm = 0.99999f;               // (1 - 1e-5)/sqrt(c)
    float sclip = (yn > max_norm) ? (max_norm / fmaxf(yn, 1e-7f)) : 1.0f;
    float yn2   = (yn > max_norm) ? (max_norm * max_norm) : yn2raw;
    float lam   = 2.0f / fmaxf(1.0f - yn2, 1e-7f);
    float rs    = ratio * sclip;                   // scale applied to raw GEMM result

    // per-column constants for this lane's two columns (n0 and 16+n0)
    float zn0 = znp[n0], zn1 = znp[16 + n0];
    float ch0 = chp[n0], ch1 = chp[16 + n0];
    float sh0 = shp[n0], sh1 = shp[16 + n0];

    #pragma unroll
    for (int rI = 0; rI < 8; ++rI) {
        // D layout: VGPR rI holds row rI (lanes 0-15) / rI+8 (lanes 16-31), col = lane%16
        int   src  = rI + (khalf << 3);            // lane owning this row's scalars
        float lamr = __shfl(lam, src);
        float rsr  = __shfl(rs,  src);

        float xz0  = rsr * acc0[rI];
        float xz1  = rsr * acc1[rI];
        float arg0 = lamr * (xz0 / zn0) * ch0 - (lamr - 1.0f) * sh0;
        float arg1 = lamr * (xz1 / zn1) * ch1 - (lamr - 1.0f) * sh1;
        float v0   = 2.0f * zn0 * asinhf(arg0);
        float v1   = 2.0f * zn1 * asinhf(arg1);
        v0 = 50.0f * tanhf(v0 * 0.02f);            // soft clamp
        v1 = 50.0f * tanhf(v1 * 0.02f);
        float w0 = sinhf(v0);                      // sinh(sqrt(c)*v)/sqrt(c), c=1
        float w1 = sinhf(v1);

        // row-wise ||w||^2 over 32 columns: reduce within each 16-lane half
        float rsum = w0 * w0 + w1 * w1;
        rsum += __shfl_xor(rsum, 1);
        rsum += __shfl_xor(rsum, 2);
        rsum += __shfl_xor(rsum, 4);
        rsum += __shfl_xor(rsum, 8);
        float inv = 1.0f / (1.0f + sqrtf(1.0f + rsum));

        int row = rowBase + rI + (khalf << 3);
        float* o = out + (size_t)row * NCOUT + n0;
        o[0]  = w0 * inv;
        o[16] = w1 * inv;
    }
}

extern "C" void kernel_launch(void* const* d_in, const int* in_sizes, int n_in,
                              void* d_out, int out_size, void* d_ws, size_t ws_size,
                              hipStream_t stream) {
    const float* x = (const float*)d_in[0];   // (64,64,64,16) f32
    const float* z = (const float*)d_in[1];   // (144,32) f32
    const float* r = (const float*)d_in[2];   // (32,) f32
    float* out = (float*)d_out;               // (64,64,64,32) f32

    hyp_prep<<<1, 64, 0, stream>>>(z, r, d_ws);
    // 262144 rows / 128 rows-per-block = 2048 blocks (8 waves x 16 rows)
    hyp_conv<<<2048, 256, 0, stream>>>(x, out, d_ws);
}
